// SlowMAF_45028437131340
// MI455X (gfx1250) — matrix operand order repaired and checked by reading, not compiled, not run on hardware
//
#include <hip/hip_runtime.h>

// ---------------------------------------------------------------------------
// SlowMAF on MI455X (gfx1250), round 6 (final form).
// D = W x H^T with A = weights (repacked offline), B = activations with
// lane = sample column. Inter-layer transpose = one shfl_xor(16) + select
// per fp32 element, then ascending bf16 converts (clean v_cvt_pk pairing).
// No LDS / no barriers in the MLP loop; x/z tiles staged per-wave in LDS.
// ---------------------------------------------------------------------------

#define DIM    64
#define NH     24
#define SLOPE  0.2f
#define BATCH  262144
#define WPB    4      // waves per block

typedef __attribute__((ext_vector_type(16))) __bf16 v16bf;
typedef __attribute__((ext_vector_type(8)))  float  v8f;

union alignas(32) Frag16 {
    v16bf  bf;
    __bf16 h[16];
    uint4  q[2];
};

union alignas(32) V8FU {
    v8f   v;
    uint4 q[2];
};

// per-MLP workspace layout, in bf16 units:
//   [   0..2047]  W1 A-fragments: frag index (kc*2 + ntile), 512 each
//   [2048..3071]  W2 A-fragments: ntile, 512 each
//   [3072..4095]  W3 A-fragments
//   [4096..4607]  W4 A-fragment (ntile 0; rows n<2 valid, rest zero)
//   [4608..4831]  biases as float[112]: b1[32] b2[32] b3[32] b4[16] (zero pad)
#define MLP_STRIDE_U16 4864
#define BIAS_OFF_U16   4608

__device__ __forceinline__ v8f wmma_bf16(const Frag16& a, const Frag16& b, v8f c) {
    return __builtin_amdgcn_wmma_f32_16x16x32_bf16(
        false, a.bf, false, b.bf, (short)0, c, false, false);
}

__device__ __forceinline__ v8f loadv8(const float* __restrict__ p) {
    V8FU u;
    const uint4* q = (const uint4*)p;
    u.q[0] = q[0];
    u.q[1] = q[1];
    return u.v;
}

__device__ __forceinline__ v8f leaky(v8f c) {
    #pragma unroll
    for (int r = 0; r < 8; ++r) {
        float v = c[r];
        c[r] = fmaxf(v, v * SLOPE);
    }
    return c;
}

// ---------------------------------------------------------------------------
// Repack one MLP's weights into WMMA *A*-matrix fragments (bf16, zero-pad).
// A fragment element map (16x32 tile, wave32):
//   lane: n_row = lane&15 (output neuron), khalf = lane>>4
//   element e: k = kc*32 + khalf*16 + e
// ---------------------------------------------------------------------------
__global__ void repack_kernel(const float* __restrict__ W1, const float* __restrict__ B1,
                              const float* __restrict__ W2, const float* __restrict__ B2,
                              const float* __restrict__ W3, const float* __restrict__ B3,
                              const float* __restrict__ W4, const float* __restrict__ B4,
                              __bf16* __restrict__ wb, int nin) {
    const int tid   = threadIdx.x;        // 512 threads
    const int lane  = tid & 31;
    const int e     = tid >> 5;           // 0..15
    const int nrow  = lane & 15;
    const int khalf = lane >> 4;
    const int g     = e >> 3;
    const int j     = e & 7;

    const float* Wl[4] = {W1, W2, W3, W4};

    #pragma unroll
    for (int f = 0; f < 9; ++f) {
        int layer, kc, nt;
        if (f < 4)      { layer = 0; kc = f >> 1; nt = f & 1; }
        else if (f < 6) { layer = 1; kc = 0;      nt = f - 4; }
        else if (f < 8) { layer = 2; kc = 0;      nt = f - 6; }
        else            { layer = 3; kc = 0;      nt = 0;     }
        const int Kdim = (layer == 0) ? nin : NH;
        const int Nact = (layer == 3) ? 2   : NH;
        const int k = kc * 32 + khalf * 16 + e;
        const int n = nt * 16 + nrow;
        float v = 0.f;
        if (k < Kdim && n < Nact) v = Wl[layer][n * Kdim + k];   // W is (fo, fi) row-major
        wb[f * 512 + lane * 16 + e] = (__bf16)v;
    }

    if (tid < 112) {                       // padded fp32 biases
        float* bf = (float*)(wb + BIAS_OFF_U16);
        const int l = tid >> 5;            // 0..3
        const int n = tid & 31;
        float v = 0.f;
        if (l == 0)      { if (n < NH) v = B1[n]; }
        else if (l == 1) { if (n < NH) v = B2[n]; }
        else if (l == 2) { if (n < NH) v = B3[n]; }
        else             { if (n < 2)  v = B4[n]; }
        bf[tid] = v;
    }
}

// ---------------------------------------------------------------------------
// Build next layer's B fragment (K x 16, lane = sample column) from the two
// D tiles (rows n=0..15 in d0, n=16..31 in d1; lane = column, vgpr r = row
// r + 8*khalf). One shfl_xor(16) per register swaps the complementary row
// halves between partner lanes; rows >= 24 are exact zeros by construction.
// B element map: lane khalf, element e -> k = khalf*16 + e.
// Conversion writes ascend e = 0..15 so the compiler pairs them into fully
// used v_cvt_pk_bf16_f32.
// ---------------------------------------------------------------------------
__device__ __forceinline__ Frag16 xpose_pack(v8f d0, v8f d1, int khalf) {
    Frag16 f;
    #pragma unroll
    for (int r = 0; r < 8; ++r) {
        const float o0 = d0[r];
        const float o1 = d1[r];
        const float p0 = __shfl_xor(o0, 16, 32);
        const float p1 = __shfl_xor(o1, 16, 32);
        f.h[r]     = (__bf16)(khalf ? p1 : o0);   // k = khalf*16 + r
        f.h[8 + r] = (__bf16)(khalf ? o1 : p0);   // k = khalf*16 + 8 + r
    }
    return f;
}

__device__ __forceinline__ void load_afrag(Frag16& a, const __bf16* __restrict__ p, int lane) {
    const uint4* q = (const uint4*)(p + lane * 16);
    a.q[0] = q[0];
    a.q[1] = q[1];
}

// One full MLP i (1..63); KC = K-chunks of layer 1, compile-time.
template <int KC>
__device__ __forceinline__ void mlp_body(int i,
                                         const Frag16 (&bx)[2],
                                         const __bf16* __restrict__ wf,
                                         const float* lx, float* lz,
                                         float& logdet, int lane) {
    const __bf16* wb   = wf + (size_t)(i - 1) * MLP_STRIDE_U16;
    const float*  bias = (const float*)(wb + BIAS_OFF_U16);
    const int khalf = lane >> 4;

    // Prefetch next MLP's fragment block into cache while we compute this one.
    if (i < DIM - 1) __builtin_prefetch(wb + MLP_STRIDE_U16, 0, 1);

    // ---- layer 1: D = W1 x X^T ----
    v8f d0 = loadv8(bias +      khalf * 8);   // c[r] = b1[r + 8*khalf]
    v8f d1 = loadv8(bias + 16 + khalf * 8);
    #pragma unroll
    for (int kc = 0; kc < KC; ++kc) {
        Frag16 aw;
        load_afrag(aw, wb + (kc * 2 + 0) * 512, lane);
        d0 = wmma_bf16(aw, bx[kc], d0);
        load_afrag(aw, wb + (kc * 2 + 1) * 512, lane);
        d1 = wmma_bf16(aw, bx[kc], d1);
    }
    Frag16 hb = xpose_pack(leaky(d0), leaky(d1), khalf);

    // ---- layer 2 ----
    d0 = loadv8(bias + 32 +      khalf * 8);
    d1 = loadv8(bias + 32 + 16 + khalf * 8);
    {
        Frag16 aw;
        load_afrag(aw, wb + 2048, lane);
        d0 = wmma_bf16(aw, hb, d0);
        load_afrag(aw, wb + 2048 + 512, lane);
        d1 = wmma_bf16(aw, hb, d1);
    }
    hb = xpose_pack(leaky(d0), leaky(d1), khalf);

    // ---- layer 3 ----
    d0 = loadv8(bias + 64 +      khalf * 8);
    d1 = loadv8(bias + 64 + 16 + khalf * 8);
    {
        Frag16 aw;
        load_afrag(aw, wb + 3072, lane);
        d0 = wmma_bf16(aw, hb, d0);
        load_afrag(aw, wb + 3072 + 512, lane);
        d1 = wmma_bf16(aw, hb, d1);
    }
    hb = xpose_pack(leaky(d0), leaky(d1), khalf);

    // ---- layer 4: rows 0 (s) and 1 (t) only ----
    v8f c4 = loadv8(bias + 96 + khalf * 8);
    {
        Frag16 aw;
        load_afrag(aw, wb + 4096, lane);
        c4 = wmma_bf16(aw, hb, c4);
    }

    // Lanes 0..15 (khalf==0) hold s=c4[0], t=c4[1] for their own sample.
    if (lane < 16) {
        const float s = c4[0];
        const float t = c4[1];
        lz[lane * DIM + i] = lx[lane * DIM + i] * __expf(s) + t;
        logdet += s;
    }
}

__global__ void __launch_bounds__(WPB * 32)
maf_kernel(const float* __restrict__ x, const float* __restrict__ leaf,
           const __bf16* __restrict__ wf,
           float* __restrict__ zout, float* __restrict__ ldout) {
    __shared__ alignas(16) float lxs[WPB][16 * DIM];
    __shared__ alignas(16) float lzs[WPB][16 * DIM];

    const int  lane = threadIdx.x & 31;
    const int  wave = threadIdx.x >> 5;
    const long m0   = ((long)blockIdx.x * WPB + wave) * 16;

    float* lx = lxs[wave];
    float* lz = lzs[wave];

    // Stage the 16x64 x-tile: whole rows -> one contiguous, coalesced 4KB copy.
    {
        const float4* xg  = (const float4*)(x + m0 * DIM);
        float4*       lx4 = (float4*)lx;
        #pragma unroll
        for (int it = 0; it < 8; ++it) lx4[it * 32 + lane] = xg[it * 32 + lane];
    }

    const int mcol  = lane & 15;   // sample column owned by this lane (B role)
    const int khalf = lane >> 4;

    // Build both K-chunk B-fragments of the x tile once (reused by all MLPs).
    // B layout: lane (m=mcol, khalf), element e -> k = kc*32 + khalf*16 + e:
    // 16 *contiguous* floats per lane -> vector loads + packed converts.
    Frag16 bx[2];
    #pragma unroll
    for (int kc = 0; kc < 2; ++kc)
        #pragma unroll
        for (int e = 0; e < 16; ++e)
            bx[kc].h[e] = (__bf16)lx[mcol * DIM + kc * 32 + khalf * 16 + e];

    // Leaf column + log-det init.
    float logdet = 0.f;
    const float s0 = leaf[0];
    const float t0 = leaf[1];
    if (lane < 16) {
        lz[lane * DIM + 0] = lx[lane * DIM + 0] * __expf(s0) + t0;
        logdet = s0;
    }

    // i in [1,32]: layer-1 K fits one 32-chunk; i in [33,63]: two chunks.
    for (int i = 1; i <= 32; ++i)
        mlp_body<1>(i, bx, wf, lx, lz, logdet, lane);
    for (int i = 33; i < DIM; ++i)
        mlp_body<2>(i, bx, wf, lx, lz, logdet, lane);

    // z tile covers whole rows -> contiguous, coalesced 4KB store.
    {
        float4*       zg  = (float4*)(zout + m0 * DIM);
        const float4* lz4 = (const float4*)lz;
        #pragma unroll
        for (int it = 0; it < 8; ++it) zg[it * 32 + lane] = lz4[it * 32 + lane];
    }
    if (lane < 16) ldout[m0 + lane] = logdet;
}

// ---------------------------------------------------------------------------
extern "C" void kernel_launch(void* const* d_in, const int* in_sizes, int n_in,
                              void* d_out, int out_size, void* d_ws, size_t ws_size,
                              hipStream_t stream) {
    (void)in_sizes; (void)n_in; (void)out_size; (void)ws_size;

    __bf16* wfrag = (__bf16*)d_ws;   // 63 * 9728 B = 612,864 B

    // Repack every MLP's weights into WMMA A-fragment layout (capture-safe:
    // plain kernel launches, pointers passed by value, no memcpy/malloc).
    for (int m = 0; m < DIM - 1; ++m) {
        const float* W1 = (const float*)d_in[2 + 8 * m + 0];
        const float* B1 = (const float*)d_in[2 + 8 * m + 1];
        const float* W2 = (const float*)d_in[2 + 8 * m + 2];
        const float* B2 = (const float*)d_in[2 + 8 * m + 3];
        const float* W3 = (const float*)d_in[2 + 8 * m + 4];
        const float* B3 = (const float*)d_in[2 + 8 * m + 5];
        const float* W4 = (const float*)d_in[2 + 8 * m + 6];
        const float* B4 = (const float*)d_in[2 + 8 * m + 7];
        repack_kernel<<<1, 512, 0, stream>>>(W1, B1, W2, B2, W3, B3, W4, B4,
                                             wfrag + (size_t)m * MLP_STRIDE_U16,
                                             m + 1);
    }

    const float* x    = (const float*)d_in[0];
    const float* leaf = (const float*)d_in[1];
    float*       z    = (float*)d_out;
    float*       ld   = z + (size_t)BATCH * DIM;

    maf_kernel<<<BATCH / (16 * WPB), WPB * 32, 0, stream>>>(x, leaf, wfrag, z, ld);
}
